// GraphNetBlock_39402029973775
// MI455X (gfx1250) — compile-verified, matrix operand order checked by reading, not run on hardware
//
#include <hip/hip_runtime.h>
#include <hip/hip_bf16.h>

// ---------------------------------------------------------------------------
// Problem constants (B, N, E, D) = (8, 2048, 16384, 128)
// ---------------------------------------------------------------------------
#define BB 8
#define NN 2048
#define EE 16384
#define DD 128

typedef __attribute__((ext_vector_type(16))) __bf16 v16bf;
typedef __attribute__((ext_vector_type(8)))  float  v8f;

// ---------------------------------------------------------------------------
// Fragment helpers.
// 16-bit A (16xK tile) / transposed-B (N-major, so identical pattern):
//   lanes 0-15 : M/N = lane,    K = {kb..kb+7} U {kb+16..kb+23}
//   lanes 16-31: M/N = lane-16, K = {kb+8..kb+15} U {kb+24..kb+31}
// -> two 16-byte contiguous loads per fragment.
// ---------------------------------------------------------------------------
__device__ __forceinline__ v16bf load_frag16(const __bf16* p) {
    union { uint4 q[2]; v16bf v; } u;
    u.q[0] = *(const uint4*)(p);
    u.q[1] = *(const uint4*)(p + 16);
    return u.v;
}

__device__ __forceinline__ const __bf16* frag_ptr(const __bf16* base, int ld,
                                                  int row, int kb, int lane) {
    return base + (size_t)row * ld + kb + ((lane >> 4) << 3);
}

__device__ __forceinline__ v8f wmma_bf16(v16bf a, v16bf b, v8f c) {
    return __builtin_amdgcn_wmma_f32_16x16x32_bf16(
        /*neg_a=*/false, a, /*neg_b=*/false, b,
        /*c_mod=*/(short)0, c, /*reuse_a=*/false, /*reuse_b=*/false);
}

// 4 M-tiles (64 rows) x one N-tile, K-loop with B-fragment register reuse:
// each loaded B fragment feeds 4 WMMAs.
// NOTE: unroll capped at 2 — full unroll makes the compiler hoist all LDS
// A-fragments (up to 512 VGPRs of live data) and spill to scratch.
template <int K>
__device__ __forceinline__ void gemm_m4(const __bf16* Ain,             // LDS, 64 x K
                                        const __bf16* __restrict__ Wt, // global, Nt x K
                                        int nt, int lane, v8f acc[4]) {
    #pragma unroll 2
    for (int kb = 0; kb < K; kb += 32) {
        v16bf bm = load_frag16(frag_ptr(Wt, K, nt * 16 + (lane & 15), kb, lane));
        #pragma unroll
        for (int mt = 0; mt < 4; ++mt) {
            v16bf a = load_frag16(frag_ptr(Ain, K, mt * 16 + (lane & 15), kb, lane));
            acc[mt] = wmma_bf16(a, bm, acc[mt]);
        }
    }
}

// ---------------------------------------------------------------------------
// Weight transpose + bf16 convert:  in[K][Nn] (f32) -> out[Nn][K] (bf16)
// ---------------------------------------------------------------------------
__global__ void transpose_to_bf16(const float* __restrict__ in,
                                  __bf16* __restrict__ out, int K, int Nn) {
    int i = blockIdx.x * blockDim.x + threadIdx.x;
    if (i < K * Nn) {
        int k = i / Nn, n = i - k * Nn;
        out[(size_t)n * K + k] = (__bf16)in[i];
    }
}

__global__ void zero_f32(float* __restrict__ p, int n) {
    int i = blockIdx.x * blockDim.x + threadIdx.x;
    if (i < n) p[i] = 0.0f;
}

// ---------------------------------------------------------------------------
// Edge MLP:  in = concat(edge, sender, receiver, global)  (K=512)
//            hidden 320 (ReLU), out 128.
// One workgroup (8 waves) = 64 edges (4 WMMA row-tiles). Fused epilogue:
// edge_out store, scatter-add into agg_e2n, reduction into agg_e2u.
// ---------------------------------------------------------------------------
__global__ __launch_bounds__(256) void edge_mlp_kernel(
    const float* __restrict__ edge_feats, const float* __restrict__ node_feats,
    const float* __restrict__ global_feats,
    const int* __restrict__ senders, const int* __restrict__ receivers,
    const __bf16* __restrict__ W1t, const float* __restrict__ b1,
    const __bf16* __restrict__ W2t, const float* __restrict__ b2,
    float* __restrict__ edge_out, float* __restrict__ agg_e2n,
    float* __restrict__ agg_e2u) {

    constexpr int KIN = 512, H = 320, OUT = 128;
    __shared__ __align__(16) __bf16 Ain[64 * KIN];   // 64 KB
    __shared__ __align__(16) __bf16 Hid[64 * H];     // 40 KB
    __shared__ int recvIdx[64];

    const int wg   = blockIdx.x;
    const int b    = wg / (EE / 64);
    const int e0   = (wg - b * (EE / 64)) * 64;
    const int t    = threadIdx.x;
    const int lane = t & 31;
    const int wave = t >> 5;

    // ---- gather + f32->bf16 convert into LDS ------------------------------
    for (int rr = t >> 4; rr < 64; rr += 16) {
        const int e  = e0 + rr;
        const int s  = senders[e];
        const int rc = receivers[e];
        if ((t & 15) == 0) recvIdx[rr] = rc;
        for (int c = (t & 15); c < KIN; c += 16) {
            float v;
            if (c < 128)      v = edge_feats[((size_t)(b * EE + e)) * DD + c];
            else if (c < 256) v = node_feats[((size_t)(b * NN + s)) * DD + (c - 128)];
            else if (c < 384) v = node_feats[((size_t)(b * NN + rc)) * DD + (c - 256)];
            else              v = global_feats[(size_t)b * DD + (c - 384)];
            Ain[rr * KIN + c] = (__bf16)v;
        }
    }
    __syncthreads();

    // ---- GEMM1: (64x512) x (512x320), ReLU -> Hid (bf16) ------------------
    for (int nt = wave; nt < H / 16; nt += 8) {
        v8f acc[4] = {};
        gemm_m4<KIN>(Ain, W1t, nt, lane, acc);
        const float bias = b1[nt * 16 + (lane & 15)];
        #pragma unroll
        for (int mt = 0; mt < 4; ++mt) {
            #pragma unroll
            for (int i = 0; i < 8; ++i) {
                float v = acc[mt][i] + bias;
                v = v > 0.0f ? v : 0.0f;
                const int row = mt * 16 + i + ((lane >> 4) << 3);
                Hid[row * H + nt * 16 + (lane & 15)] = (__bf16)v;
            }
        }
    }
    __syncthreads();

    // ---- GEMM2: (64x320) x (320x128) + fused epilogue ---------------------
    {
        const int nt = wave;                      // 8 waves, 8 output tiles
        v8f acc[4] = {};
        gemm_m4<H>(Hid, W2t, nt, lane, acc);
        const int   col  = nt * 16 + (lane & 15);
        const float bias = b2[col];
        float colsum = 0.0f;
        #pragma unroll
        for (int mt = 0; mt < 4; ++mt) {
            #pragma unroll
            for (int i = 0; i < 8; ++i) {
                const float v   = acc[mt][i] + bias;
                const int   row = mt * 16 + i + ((lane >> 4) << 3);
                edge_out[((size_t)(b * EE + e0 + row)) * OUT + col] = v;
                atomicAdd(&agg_e2n[((size_t)(b * NN + recvIdx[row])) * OUT + col], v);
                colsum += v;
            }
        }
        colsum += __shfl_xor(colsum, 16, 32);     // combine M lower/upper halves
        if (lane < 16) atomicAdd(&agg_e2u[b * OUT + col], colsum);
    }
}

// ---------------------------------------------------------------------------
// Node MLP: in = concat(agg_e2n, node, global) (K=384), hidden 256, out 128.
// One workgroup = 64 nodes.
// ---------------------------------------------------------------------------
__global__ __launch_bounds__(256) void node_mlp_kernel(
    const float* __restrict__ agg_e2n, const float* __restrict__ node_feats,
    const float* __restrict__ global_feats,
    const __bf16* __restrict__ W1t, const float* __restrict__ b1,
    const __bf16* __restrict__ W2t, const float* __restrict__ b2,
    float* __restrict__ node_out, float* __restrict__ agg_n2u) {

    constexpr int KIN = 384, H = 256, OUT = 128;
    __shared__ __align__(16) __bf16 Ain[64 * KIN];   // 48 KB
    __shared__ __align__(16) __bf16 Hid[64 * H];     // 32 KB

    const int wg   = blockIdx.x;
    const int b    = wg / (NN / 64);
    const int n0   = (wg - b * (NN / 64)) * 64;
    const int t    = threadIdx.x;
    const int lane = t & 31;
    const int wave = t >> 5;

    for (int rr = t >> 4; rr < 64; rr += 16) {
        const int nd = n0 + rr;
        for (int c = (t & 15); c < KIN; c += 16) {
            float v;
            if (c < 128)      v = agg_e2n[((size_t)(b * NN + nd)) * DD + c];
            else if (c < 256) v = node_feats[((size_t)(b * NN + nd)) * DD + (c - 128)];
            else              v = global_feats[(size_t)b * DD + (c - 256)];
            Ain[rr * KIN + c] = (__bf16)v;
        }
    }
    __syncthreads();

    for (int nt = wave; nt < H / 16; nt += 8) {
        v8f acc[4] = {};
        gemm_m4<KIN>(Ain, W1t, nt, lane, acc);
        const float bias = b1[nt * 16 + (lane & 15)];
        #pragma unroll
        for (int mt = 0; mt < 4; ++mt) {
            #pragma unroll
            for (int i = 0; i < 8; ++i) {
                float v = acc[mt][i] + bias;
                v = v > 0.0f ? v : 0.0f;
                const int row = mt * 16 + i + ((lane >> 4) << 3);
                Hid[row * H + nt * 16 + (lane & 15)] = (__bf16)v;
            }
        }
    }
    __syncthreads();

    {
        const int nt = wave;
        v8f acc[4] = {};
        gemm_m4<H>(Hid, W2t, nt, lane, acc);
        const int   col  = nt * 16 + (lane & 15);
        const float bias = b2[col];
        float colsum = 0.0f;
        #pragma unroll
        for (int mt = 0; mt < 4; ++mt) {
            #pragma unroll
            for (int i = 0; i < 8; ++i) {
                const float v   = acc[mt][i] + bias;
                const int   row = mt * 16 + i + ((lane >> 4) << 3);
                node_out[((size_t)(b * NN + n0 + row)) * OUT + col] = v;
                colsum += v;
            }
        }
        colsum += __shfl_xor(colsum, 16, 32);
        if (lane < 16) atomicAdd(&agg_n2u[b * OUT + col], colsum);
    }
}

// ---------------------------------------------------------------------------
// Global MLP: 8 rows (padded to one 16-row WMMA tile), K=384, H=256, out 128.
// ---------------------------------------------------------------------------
__global__ __launch_bounds__(256) void glob_mlp_kernel(
    const float* __restrict__ agg_e2u, const float* __restrict__ agg_n2u,
    const float* __restrict__ global_feats,
    const __bf16* __restrict__ W1t, const float* __restrict__ b1,
    const __bf16* __restrict__ W2t, const float* __restrict__ b2,
    float* __restrict__ glob_out) {

    constexpr int KIN = 384, H = 256;
    __shared__ __align__(16) __bf16 Ain[16 * KIN];
    __shared__ __align__(16) __bf16 Hid[16 * H];

    const int t    = threadIdx.x;
    const int lane = t & 31;
    const int wave = t >> 5;

    {
        const int r = t >> 4;                 // row == batch index, rows 8-15 pad
        for (int c = (t & 15); c < KIN; c += 16) {
            float v = 0.0f;
            if (r < BB) {
                if (c < 128)      v = agg_e2u[r * DD + c];
                else if (c < 256) v = agg_n2u[r * DD + (c - 128)];
                else              v = global_feats[r * DD + (c - 256)];
            }
            Ain[r * KIN + c] = (__bf16)v;
        }
    }
    __syncthreads();

    for (int nt = wave; nt < H / 16; nt += 8) {
        v8f acc = {};
        #pragma unroll 2
        for (int kb = 0; kb < KIN; kb += 32) {
            v16bf a  = load_frag16(frag_ptr(Ain, KIN, lane & 15, kb, lane));
            v16bf bm = load_frag16(frag_ptr(W1t, KIN, nt * 16 + (lane & 15), kb, lane));
            acc = wmma_bf16(a, bm, acc);
        }
        const float bias = b1[nt * 16 + (lane & 15)];
        #pragma unroll
        for (int i = 0; i < 8; ++i) {
            float v = acc[i] + bias;
            v = v > 0.0f ? v : 0.0f;
            const int row = i + ((lane >> 4) << 3);
            Hid[row * H + nt * 16 + (lane & 15)] = (__bf16)v;
        }
    }
    __syncthreads();

    {
        const int nt = wave;
        v8f acc = {};
        #pragma unroll 2
        for (int kb = 0; kb < H; kb += 32) {
            v16bf a  = load_frag16(frag_ptr(Hid, H, lane & 15, kb, lane));
            v16bf bm = load_frag16(frag_ptr(W2t, H, nt * 16 + (lane & 15), kb, lane));
            acc = wmma_bf16(a, bm, acc);
        }
        const int   col  = nt * 16 + (lane & 15);
        const float bias = b2[col];
        #pragma unroll
        for (int i = 0; i < 8; ++i) {
            const int row = i + ((lane >> 4) << 3);
            if (row < BB) glob_out[row * DD + col] = acc[i] + bias;
        }
    }
}

// ---------------------------------------------------------------------------
// Host-side orchestration
// ---------------------------------------------------------------------------
extern "C" void kernel_launch(void* const* d_in, const int* in_sizes, int n_in,
                              void* d_out, int out_size, void* d_ws, size_t ws_size,
                              hipStream_t stream) {
    const float* edge_feats   = (const float*)d_in[0];
    const float* node_feats   = (const float*)d_in[1];
    const float* global_feats = (const float*)d_in[2];
    const int*   senders      = (const int*)d_in[3];
    const int*   receivers    = (const int*)d_in[4];
    const float* We1 = (const float*)d_in[5];  const float* be1 = (const float*)d_in[6];
    const float* We2 = (const float*)d_in[7];  const float* be2 = (const float*)d_in[8];
    const float* Wn1 = (const float*)d_in[9];  const float* bn1 = (const float*)d_in[10];
    const float* Wn2 = (const float*)d_in[11]; const float* bn2 = (const float*)d_in[12];
    const float* Wu1 = (const float*)d_in[13]; const float* bu1 = (const float*)d_in[14];
    const float* Wu2 = (const float*)d_in[15]; const float* bu2 = (const float*)d_in[16];

    // Workspace layout (all offsets multiple of 256 bytes)
    char* ws = (char*)d_ws;
    size_t off = 0;
    __bf16* We1t = (__bf16*)(ws + off); off += (size_t)512 * 320 * 2;  // 327680
    __bf16* We2t = (__bf16*)(ws + off); off += (size_t)320 * 128 * 2;  //  81920
    __bf16* Wn1t = (__bf16*)(ws + off); off += (size_t)384 * 256 * 2;  // 196608
    __bf16* Wn2t = (__bf16*)(ws + off); off += (size_t)256 * 128 * 2;  //  65536
    __bf16* Wu1t = (__bf16*)(ws + off); off += (size_t)384 * 256 * 2;  // 196608
    __bf16* Wu2t = (__bf16*)(ws + off); off += (size_t)256 * 128 * 2;  //  65536
    float* agg_e2n = (float*)(ws + off); off += (size_t)BB * NN * DD * 4;
    float* agg_e2u = (float*)(ws + off); off += (size_t)BB * DD * 4;
    float* agg_n2u = (float*)(ws + off); off += (size_t)BB * DD * 4;

    float* edge_out = (float*)d_out;
    float* node_out = edge_out + (size_t)BB * EE * DD;
    float* glob_out = node_out + (size_t)BB * NN * DD;

    const dim3 blk(256);

    // Zero the accumulators (agg_e2n, agg_e2u, agg_n2u are contiguous)
    {
        const int nz = BB * NN * DD + 2 * BB * DD;
        zero_f32<<<(nz + 255) / 256, blk, 0, stream>>>(agg_e2n, nz);
    }
    // Weight transpose + bf16 convert
    transpose_to_bf16<<<(512 * 320 + 255) / 256, blk, 0, stream>>>(We1, We1t, 512, 320);
    transpose_to_bf16<<<(320 * 128 + 255) / 256, blk, 0, stream>>>(We2, We2t, 320, 128);
    transpose_to_bf16<<<(384 * 256 + 255) / 256, blk, 0, stream>>>(Wn1, Wn1t, 384, 256);
    transpose_to_bf16<<<(256 * 128 + 255) / 256, blk, 0, stream>>>(Wn2, Wn2t, 256, 128);
    transpose_to_bf16<<<(384 * 256 + 255) / 256, blk, 0, stream>>>(Wu1, Wu1t, 384, 256);
    transpose_to_bf16<<<(256 * 128 + 255) / 256, blk, 0, stream>>>(Wu2, Wu2t, 256, 128);

    // Edge MLP + fused scatter/reductions   (64 edges per workgroup)
    edge_mlp_kernel<<<BB * (EE / 64), blk, 0, stream>>>(
        edge_feats, node_feats, global_feats, senders, receivers,
        We1t, be1, We2t, be2, edge_out, agg_e2n, agg_e2u);

    // Node MLP + fused reduction            (64 nodes per workgroup)
    node_mlp_kernel<<<BB * (NN / 64), blk, 0, stream>>>(
        agg_e2n, node_feats, global_feats, Wn1t, bn1, Wn2t, bn2,
        node_out, agg_n2u);

    // Global MLP
    glob_mlp_kernel<<<1, blk, 0, stream>>>(
        agg_e2u, agg_n2u, global_feats, Wu1t, bu1, Wu2t, bu2, glob_out);
}
